// CausalSelfAttention_20194936226578
// MI455X (gfx1250) — matrix operand order repaired and checked by reading, not compiled, not run on hardware
//
#include <hip/hip_runtime.h>

// ---------------------------------------------------------------------------
// CDNA5 (gfx1250) causal self-attention: bf16 WMMA pipeline, wave32,
// async LDS staging (ASYNCcnt) with double buffering.
// ---------------------------------------------------------------------------

typedef __bf16 bf16;
typedef __attribute__((ext_vector_type(16))) __bf16 v16bf;
typedef __attribute__((ext_vector_type(8)))  __bf16 v8bf;
typedef __attribute__((ext_vector_type(8)))  float  v8f;
typedef int v4i __attribute__((vector_size(16)));   // matches builtin param pointee

union frag16 { v16bf v; v8bf h[2]; };

#define GLOBAL_AS __attribute__((address_space(1)))
#define LDS_AS    __attribute__((address_space(3)))

#if __has_builtin(__builtin_amdgcn_global_load_async_to_lds_b128) && \
    __has_builtin(__builtin_amdgcn_s_wait_asynccnt)
#define HAVE_ASYNC 1
#endif

// 16-byte global -> LDS copy. Async (ASYNCcnt-tracked) when available.
__device__ __forceinline__ void cp_b128(bf16* lds_dst, const bf16* gsrc) {
#ifdef HAVE_ASYNC
  __builtin_amdgcn_global_load_async_to_lds_b128(
      (GLOBAL_AS v4i*)(GLOBAL_AS void*)(void*)gsrc,
      (LDS_AS v4i*)(LDS_AS void*)(void*)lds_dst,
      /*offset=*/0, /*cpol=*/0);
#else
  *(v8bf*)lds_dst = *(const v8bf*)gsrc;
#endif
}

template <int N>
__device__ __forceinline__ void wait_async() {
#ifdef HAVE_ASYNC
  __builtin_amdgcn_s_wait_asynccnt(N);
#endif
}

__device__ __forceinline__ v8f wmma_bf16(v16bf a, v16bf b, v8f c) {
  // D = A(16x32 bf16) * B(32x16 bf16) + C(16x16 f32)
  return __builtin_amdgcn_wmma_f32_16x16x32_bf16(
      /*neg_a=*/false, a, /*neg_b=*/false, b,
      /*c_mod=*/(short)0, c, /*reuse_a=*/false, /*reuse_b=*/false);
}

// Problem constants
constexpr int B_   = 4;
constexpr int T_   = 2048;
constexpr int C_   = 1024;
constexpr int H_   = 16;
constexpr int HD_  = 64;
constexpr int QKVN = 3 * C_;   // 3072
constexpr int M_   = B_ * T_;  // 8192

// ---------------------------------------------------------------------------
// f32 -> bf16 elementwise convert
// ---------------------------------------------------------------------------
__global__ __launch_bounds__(256) void cvt_f32_bf16(const float* __restrict__ in,
                                                    bf16* __restrict__ out,
                                                    long n) {
  long i = (long)blockIdx.x * blockDim.x + threadIdx.x;
  if (i < n) out[i] = (bf16)in[i];
}

// ---------------------------------------------------------------------------
// f32 [K][N] -> bf16 [N][K]  (transpose + convert; coalesced reads)
// ---------------------------------------------------------------------------
__global__ __launch_bounds__(256) void trcvt_f32_bf16(const float* __restrict__ in,
                                                      bf16* __restrict__ out,
                                                      int K, int N) {
  long gid = (long)blockIdx.x * blockDim.x + threadIdx.x;
  if (gid < (long)K * N) {
    int k = (int)(gid / N);
    int n = (int)(gid - (long)k * N);
    out[(long)n * K + k] = (bf16)in[gid];
  }
}

// ---------------------------------------------------------------------------
// bf16 WMMA GEMM: OUT[M][N] = A[M][K] * BT[N][K]^T + bias[N]
// Block tile 64(M) x 128(N), 8 waves (4 M x 2 N), wave tile 16x64.
// Double-buffered async LDS staging of A (64x32) and B (128x32) K-slices.
// ---------------------------------------------------------------------------
template <bool OUT_F32>
__global__ __launch_bounds__(256) void gemm_bf16(const bf16* __restrict__ A,
                                                 const bf16* __restrict__ BT,
                                                 const float* __restrict__ bias,
                                                 float* __restrict__ outf,
                                                 bf16* __restrict__ outb,
                                                 int M, int N, int K) {
  __shared__ bf16 lsA[2][64 * 32];    // [row][k]  4 KB / buffer
  __shared__ bf16 lsB[2][128 * 32];   // [col][k]  8 KB / buffer

  const int tid  = threadIdx.x;
  const int lane = tid & 31;
  const int wv   = tid >> 5;
  const int col  = lane & 15;
  const int hh   = lane >> 4;
  const int wm   = wv & 3;
  const int wn   = wv >> 2;

  const int mblk = blockIdx.y * 64;
  const int nblk = blockIdx.x * 128;

  v8f zero = {};
  v8f acc[4] = {zero, zero, zero, zero};

  // Stage one 32-wide K slice into buffer `buf` (3 x 16B async ops / thread).
  auto stage = [&](int buf, int k0) {
    {
      const int row = tid >> 2;          // 0..63
      const int ko  = (tid & 3) * 8;     // 0,8,16,24
      cp_b128(&lsA[buf][row * 32 + ko], A + (size_t)(mblk + row) * K + k0 + ko);
    }
#pragma unroll
    for (int i = 0; i < 2; ++i) {
      const int c  = tid + i * 256;      // 0..511
      const int n  = c >> 2;             // 0..127
      const int ko = (c & 3) * 8;
      cp_b128(&lsB[buf][n * 32 + ko], BT + (size_t)(nblk + n) * K + k0 + ko);
    }
  };

  const int NT = K >> 5;                 // # of 32-wide K slices
  stage(0, 0);

  for (int it = 0; it < NT; ++it) {
    const int cur = it & 1;
    if (it + 1 < NT) {                   // overlap: issue next slice first
      stage(cur ^ 1, (it + 1) * 32);
      wait_async<3>();                   // allow the 3 just-issued ops in flight
    } else {
      wait_async<0>();
    }
    __syncthreads();                     // slice `cur` fully resident

    const bf16* As = lsA[cur];
    const bf16* Bs = lsB[cur];
    frag16 ua;
    ua.h[0] = *(const v8bf*)(As + (wm * 16 + col) * 32 + hh * 16);
    ua.h[1] = *(const v8bf*)(As + (wm * 16 + col) * 32 + hh * 16 + 8);
#pragma unroll
    for (int j = 0; j < 4; ++j) {
      const bf16* bp = Bs + (wn * 64 + j * 16 + col) * 32 + hh * 16;
      frag16 ub;
      ub.h[0] = *(const v8bf*)(bp);
      ub.h[1] = *(const v8bf*)(bp + 8);
      acc[j] = wmma_bf16(ua.v, ub.v, acc[j]);
    }
    __syncthreads();                     // everyone done before overwrite
  }

#pragma unroll
  for (int j = 0; j < 4; ++j) {
    const int n  = nblk + wn * 64 + j * 16 + col;
    const float bn = bias ? bias[n] : 0.0f;
#pragma unroll
    for (int r = 0; r < 8; ++r) {
      const size_t row = (size_t)(mblk + wm * 16 + r + 8 * hh);
      const float v = acc[j][r] + bn;
      if (OUT_F32) outf[row * N + n] = v;
      else         outb[row * N + n] = (bf16)v;
    }
  }
}

// ---------------------------------------------------------------------------
// Flash-style causal attention. qkv: bf16 [B*T][3072]; per head h:
//   Q col off = h*64 ; K col off = 1024 + h*64 ; V col off = 2048 + h*64.
// grid = (T/128, H, B), block = 256 (8 waves; wave w owns rows qbase+16w..+15).
// Output: bf16 [B*T][C].
// ---------------------------------------------------------------------------
__global__ __launch_bounds__(256) void attn_causal(const bf16* __restrict__ qkv,
                                                   bf16* __restrict__ obuf) {
  __shared__ bf16 lsK[32 * 64];        // [key][hd], row-major
  __shared__ bf16 lsVt[64 * 40];       // [hd][key], stride 40 (16B aligned rows)
  __shared__ bf16 lsP[8 * 16 * 32];    // per-wave P tile [row][key]

  const int tid  = threadIdx.x;
  const int lane = tid & 31;
  const int wv   = tid >> 5;
  const int col  = lane & 15;
  const int hh   = lane >> 4;

  const int qt = blockIdx.x;
  const int h  = blockIdx.y;
  const int b  = blockIdx.z;
  const size_t rowbase = (size_t)b * T_;
  const int qbase = qt * 128;
  const int q0    = qbase + wv * 16;

  // --- load this wave's Q fragments (16 rows x 64 hd, two K=32 chunks) ---
  const bf16* qptr = qkv + (rowbase + q0 + col) * QKVN + h * HD_;
  frag16 aq[2];
#pragma unroll
  for (int kc = 0; kc < 2; ++kc) {
    aq[kc].h[0] = *(const v8bf*)(qptr + kc * 32 + hh * 16);
    aq[kc].h[1] = *(const v8bf*)(qptr + kc * 32 + hh * 16 + 8);
  }

  v8f zero = {};
  v8f o[4] = {zero, zero, zero, zero};   // 16 x 64 output accumulator
  float mrow[8], lrow[8];
#pragma unroll
  for (int r = 0; r < 8; ++r) { mrow[r] = -__builtin_inff(); lrow[r] = 0.0f; }

  const float scale = 0.125f;            // 1/sqrt(64)
  const int nk = (qbase + 128) >> 5;     // 32-key tiles needed (causal bound)

  for (int it = 0; it < nk; ++it) {
    const int kbase = it * 32;

    // --- cooperative stage: K tile async (row-major), V tile transposed ---
    {
      const int key = tid >> 3;          // 0..31
      const int hdo = (tid & 7) * 8;     // 0..56
      const bf16* kp = qkv + (rowbase + kbase + key) * QKVN + C_ + h * HD_ + hdo;
      cp_b128(&lsK[key * 64 + hdo], kp);
      const bf16* vp = qkv + (rowbase + kbase + key) * QKVN + 2 * C_ + h * HD_ + hdo;
      v8bf vv = *(const v8bf*)vp;
#pragma unroll
      for (int i = 0; i < 8; ++i) lsVt[(hdo + i) * 40 + key] = vv[i];
    }
    wait_async<0>();
    __syncthreads();

    // --- S = Q @ K^T : two 16-key subtiles, K-dim (hd) = 64 in 2 steps ---
    v8f s[2];
#pragma unroll
    for (int nt = 0; nt < 2; ++nt) {
      v8f a = zero;
#pragma unroll
      for (int kc = 0; kc < 2; ++kc) {
        const bf16* bp = &lsK[(nt * 16 + col) * 64 + kc * 32 + hh * 16];
        frag16 ub;
        ub.h[0] = *(const v8bf*)(bp);
        ub.h[1] = *(const v8bf*)(bp + 8);
        a = wmma_bf16(aq[kc].v, ub.v, a);
      }
      s[nt] = a;
    }

    // --- online softmax over the 32 keys ---
#pragma unroll
    for (int r = 0; r < 8; ++r) {
      const int qrow = q0 + r + 8 * hh;
      float s0 = s[0][r] * scale;
      float s1 = s[1][r] * scale;
      if (kbase + col > qrow)      s0 = -__builtin_inff();
      if (kbase + 16 + col > qrow) s1 = -__builtin_inff();
      float rmax = fmaxf(s0, s1);
#pragma unroll
      for (int m = 8; m >= 1; m >>= 1) rmax = fmaxf(rmax, __shfl_xor(rmax, m));
      const float mnew = fmaxf(mrow[r], rmax);
      float alpha, p0, p1;
      if (mnew == -__builtin_inff()) {   // fully masked so far
        alpha = 1.0f; p0 = 0.0f; p1 = 0.0f;
      } else {
        alpha = __expf(mrow[r] - mnew);  // exp(-inf)=0 on first hit
        p0 = __expf(s0 - mnew);
        p1 = __expf(s1 - mnew);
      }
      float ls = p0 + p1;
#pragma unroll
      for (int m = 8; m >= 1; m >>= 1) ls += __shfl_xor(ls, m);
      lrow[r] = lrow[r] * alpha + ls;
      mrow[r] = mnew;
#pragma unroll
      for (int j = 0; j < 4; ++j) o[j][r] *= alpha;
      bf16* pp = &lsP[wv * 512 + (r + 8 * hh) * 32];
      pp[col]      = (bf16)p0;
      pp[16 + col] = (bf16)p1;
    }
    __syncthreads();  // order P stores vs. A-fragment loads; hold lsK/lsVt

    // --- O += P(16x32) @ V(32x64) ---
    frag16 ua;
    {
      const bf16* pb = &lsP[wv * 512 + col * 32];
      ua.h[0] = *(const v8bf*)(pb + hh * 16);
      ua.h[1] = *(const v8bf*)(pb + hh * 16 + 8);
    }
#pragma unroll
    for (int j = 0; j < 4; ++j) {
      const bf16* vb = &lsVt[(j * 16 + col) * 40 + hh * 16];
      frag16 ub;
      ub.h[0] = *(const v8bf*)(vb);
      ub.h[1] = *(const v8bf*)(vb + 8);
      o[j] = wmma_bf16(ua.v, ub.v, o[j]);
    }
    __syncthreads();  // done with lsK / lsVt for this tile
  }

  // --- epilogue: normalize and store bf16 [B*T][C] ---
#pragma unroll
  for (int r = 0; r < 8; ++r) {
    const float inv = 1.0f / lrow[r];
    bf16* op = obuf + (rowbase + q0 + r + 8 * hh) * C_ + h * HD_;
#pragma unroll
    for (int j = 0; j < 4; ++j) op[j * 16 + col] = (bf16)(o[j][r] * inv);
  }
}

// ---------------------------------------------------------------------------
// Host-side launch
// ---------------------------------------------------------------------------
extern "C" void kernel_launch(void* const* d_in, const int* in_sizes, int n_in,
                              void* d_out, int out_size, void* d_ws, size_t ws_size,
                              hipStream_t stream) {
  (void)in_sizes; (void)n_in; (void)out_size; (void)ws_size;

  const float* x     = (const float*)d_in[0];  // [B,T,C]
  const float* Wqkv  = (const float*)d_in[1];  // [C,3C]
  const float* bqkv  = (const float*)d_in[2];  // [3C]
  const float* Wproj = (const float*)d_in[3];  // [C,C]
  const float* bproj = (const float*)d_in[4];  // [C]
  float* out = (float*)d_out;                  // [B,T,C]

  char* ws = (char*)d_ws;
  const size_t MB = 1024ull * 1024ull;
  bf16* xb     = (bf16*)(ws);              // 8192*1024  bf16  (16 MB)
  bf16* qkvb   = (bf16*)(ws + 16 * MB);    // 8192*3072  bf16  (48 MB)
  bf16* wqkvT  = (bf16*)(ws + 64 * MB);    // 3072*1024  bf16  ( 6 MB)
  bf16* wprojT = (bf16*)(ws + 70 * MB);    // 1024*1024  bf16  ( 2 MB)
  bf16* obuf   = (bf16*)(ws + 72 * MB);    // 8192*1024  bf16  (16 MB)

  // 1) convert activations / weights to bf16 (weights transposed to [N][K])
  {
    long n = (long)M_ * C_;
    cvt_f32_bf16<<<dim3((unsigned)((n + 255) / 256)), dim3(256), 0, stream>>>(x, xb, n);
  }
  {
    long n = (long)C_ * QKVN;
    trcvt_f32_bf16<<<dim3((unsigned)((n + 255) / 256)), dim3(256), 0, stream>>>(Wqkv, wqkvT, C_, QKVN);
  }
  {
    long n = (long)C_ * C_;
    trcvt_f32_bf16<<<dim3((unsigned)((n + 255) / 256)), dim3(256), 0, stream>>>(Wproj, wprojT, C_, C_);
  }

  // 2) QKV GEMM: qkvb[M][3C] = xb @ wqkvT^T + bqkv  (bf16 out)
  gemm_bf16<false><<<dim3(QKVN / 128, M_ / 64), dim3(256), 0, stream>>>(
      xb, wqkvT, bqkv, nullptr, qkvb, M_, QKVN, C_);

  // 3) causal flash attention -> obuf[M][C] bf16
  attn_causal<<<dim3(T_ / 128, H_, B_), dim3(256), 0, stream>>>(qkvb, obuf);

  // 4) projection GEMM: out[M][C] = obuf @ wprojT^T + bproj  (f32 out)
  gemm_bf16<true><<<dim3(C_ / 128, M_ / 64), dim3(256), 0, stream>>>(
      obuf, wprojT, bproj, out, nullptr, M_, C_, C_);
}
